// ScaledDotProductAttention_11793980195164
// MI455X (gfx1250) — matrix operand order repaired and checked by reading, not compiled, not run on hardware
//
#include <hip/hip_runtime.h>

typedef __attribute__((ext_vector_type(16))) _Float16 v16h;
typedef __attribute__((ext_vector_type(8)))  _Float16 v8h;
typedef __attribute__((ext_vector_type(4)))  _Float16 v4h;
typedef __attribute__((ext_vector_type(2)))  _Float16 v2h;
typedef __attribute__((ext_vector_type(8)))  float    v8f;

#define DKK   128
#define SEQ   2048
#define TK    64          // keys per tile
#define NKT   (SEQ / TK)  // 32 key tiles
#define WAVES 8
#define KSTR  136         // sK row stride in halves (128 + 8 pad, 16B-aligned)
#define VSTR  72          // sVt row stride in halves (64 + 8 pad, 16B-aligned)
#define PSTR  72          // sP  row stride in halves

union V16U { v16h v; v8h h[2]; v2h d[8]; _Float16 e[16]; };
union V4U  { v4h v; v2h d[2]; };

// v_cvt_pk_rtz_f16_f32: convert two f32 -> packed f16 in one VALU op
__device__ __forceinline__ v2h pk2h(float a, float b) {
    return __builtin_bit_cast(v2h, __builtin_amdgcn_cvt_pkrtz(a, b));
}

// XOR-swap across lanes via DS_SWIZZLE_B32 (group-of-32 mode):
// offset = { xor_mask[14:10], or_mask[9:5]=0, and_mask[4:0]=0x1f }
template <int XM>
__device__ __forceinline__ float swz_xor(float x) {
    int i = __builtin_bit_cast(int, x);
    i = __builtin_amdgcn_ds_swizzle(i, (XM << 10) | 0x1f);
    return __builtin_bit_cast(float, i);
}
__device__ __forceinline__ float rowmax16(float x) {
    x = fmaxf(x, swz_xor<1>(x));
    x = fmaxf(x, swz_xor<2>(x));
    x = fmaxf(x, swz_xor<4>(x));
    x = fmaxf(x, swz_xor<8>(x));
    return x;
}
__device__ __forceinline__ float rowsum16(float x) {
    x += swz_xor<1>(x);
    x += swz_xor<2>(x);
    x += swz_xor<4>(x);
    x += swz_xor<8>(x);
    return x;
}

__global__ __launch_bounds__(256)
void flash_attn_f16wmma(const float* __restrict__ Q,
                        const float* __restrict__ K,
                        const float* __restrict__ V,
                        float* __restrict__ Out)
{
    __shared__ _Float16 sK [TK   * KSTR];       // K tile, [key][d], f16
    __shared__ _Float16 sVt[DKK  * VSTR];       // V tile transposed, [d][key], f16
    __shared__ _Float16 sP [WAVES * 16 * PSTR]; // per-wave P tile, [m][k], f16

    const int tid  = threadIdx.x;
    const int wave = tid >> 5;
    const int lane = tid & 31;
    const int g    = lane >> 4;     // half-wave group
    const int l16  = lane & 15;

    const int batch = blockIdx.x >> 4;   // 16 blocks per batch
    const int qblk  = blockIdx.x & 15;
    const int q0    = qblk * 128 + wave * 16;   // first query row of this wave

    const size_t base = (size_t)batch * SEQ * DKK;

    // ---- Q fragments (A-layout), pre-scaled by 1/sqrt(d_k) ----
    const float scale = 0.088388347648318447f; // 1/sqrt(128)
    v16h qf[4];
    {
        const float* qrow = Q + base + (size_t)(q0 + l16) * DKK;
#pragma unroll
        for (int c = 0; c < 4; ++c) {
            const float4 a0 = *(const float4*)(qrow + c * 32 + g * 8);
            const float4 a1 = *(const float4*)(qrow + c * 32 + g * 8 + 4);
            const float4 a2 = *(const float4*)(qrow + c * 32 + 16 + g * 8);
            const float4 a3 = *(const float4*)(qrow + c * 32 + 16 + g * 8 + 4);
            V16U u;
            u.d[0] = pk2h(a0.x * scale, a0.y * scale);
            u.d[1] = pk2h(a0.z * scale, a0.w * scale);
            u.d[2] = pk2h(a1.x * scale, a1.y * scale);
            u.d[3] = pk2h(a1.z * scale, a1.w * scale);
            u.d[4] = pk2h(a2.x * scale, a2.y * scale);
            u.d[5] = pk2h(a2.z * scale, a2.w * scale);
            u.d[6] = pk2h(a3.x * scale, a3.y * scale);
            u.d[7] = pk2h(a3.z * scale, a3.w * scale);
            qf[c] = u.v;
        }
    }

    // ---- accumulators ----
    v8f Oacc[8];
#pragma unroll
    for (int t = 0; t < 8; ++t) { v8f z{}; Oacc[t] = z; }
    float mrow[8], lrow[8];
#pragma unroll
    for (int vv = 0; vv < 8; ++vv) { mrow[vv] = -3.0e38f; lrow[vv] = 0.0f; }

    _Float16* pw = &sP[wave * 16 * PSTR];

    // staging coordinates: 256 threads -> 64 rows x 4 col-chunks of 32
    const int srow = tid >> 2;        // 0..63 key row
    const int scol = (tid & 3) * 32;  // 0..96 d column start

    for (int kt = 0; kt < NKT; ++kt) {
        // ---- cooperative stage: K row-major f16, V transposed f16 ----
        {
            const float* krow = K + base + (size_t)(kt * TK + srow) * DKK + scol;
            const float* vrow = V + base + (size_t)(kt * TK + srow) * DKK + scol;
#pragma unroll
            for (int j = 0; j < 8; ++j) {
                const float4 kv = *(const float4*)(krow + j * 4);
                V4U p;
                p.d[0] = pk2h(kv.x, kv.y);
                p.d[1] = pk2h(kv.z, kv.w);
                *(v4h*)&sK[srow * KSTR + scol + j * 4] = p.v;

                const float4 vv4 = *(const float4*)(vrow + j * 4);
                sVt[(scol + j * 4 + 0) * VSTR + srow] = (_Float16)vv4.x;
                sVt[(scol + j * 4 + 1) * VSTR + srow] = (_Float16)vv4.y;
                sVt[(scol + j * 4 + 2) * VSTR + srow] = (_Float16)vv4.z;
                sVt[(scol + j * 4 + 3) * VSTR + srow] = (_Float16)vv4.w;
            }
        }
        __syncthreads();

        // ---- S = Q*K^T : 16 queries x 64 keys (four 16x16 tiles) ----
        // Load 4 B fragments per d-chunk into a register window, then 4 WMMAs,
        // so 8 ds_load_b128 can stay in flight per WMMA burst.
        v8f S[4];
#pragma unroll
        for (int j = 0; j < 4; ++j) { v8f z{}; S[j] = z; }
#pragma unroll
        for (int c = 0; c < 4; ++c) {
            V16U b[4];
#pragma unroll
            for (int j = 0; j < 4; ++j) {
                const _Float16* p = &sK[(j * 16 + l16) * KSTR + c * 32 + g * 8];
                b[j].h[0] = *(const v8h*)p;
                b[j].h[1] = *(const v8h*)(p + 16);
            }
#pragma unroll
            for (int j = 0; j < 4; ++j) {
                S[j] = __builtin_amdgcn_wmma_f32_16x16x32_f16(false, qf[c], false, b[j].v,
                                                              (short)0, S[j], false, false);
            }
        }

        // ---- online softmax; write P tile (f16) to per-wave LDS ----
#pragma unroll
        for (int vv = 0; vv < 8; ++vv) {
            const float s0 = S[0][vv], s1 = S[1][vv], s2 = S[2][vv], s3 = S[3][vv];
            float t = fmaxf(fmaxf(s0, s1), fmaxf(s2, s3));
            t = rowmax16(t);
            const float mn   = fmaxf(mrow[vv], t);
            const float corr = __expf(mrow[vv] - mn);
            const float p0 = __expf(s0 - mn);
            const float p1 = __expf(s1 - mn);
            const float p2 = __expf(s2 - mn);
            const float p3 = __expf(s3 - mn);
            const float rs = rowsum16((p0 + p1) + (p2 + p3));
            lrow[vv] = lrow[vv] * corr + rs;
            mrow[vv] = mn;
#pragma unroll
            for (int dt = 0; dt < 8; ++dt) Oacc[dt][vv] *= corr;
            const int r = vv + 8 * g;
            pw[r * PSTR + l16]      = (_Float16)p0;
            pw[r * PSTR + l16 + 16] = (_Float16)p1;
            pw[r * PSTR + l16 + 32] = (_Float16)p2;
            pw[r * PSTR + l16 + 48] = (_Float16)p3;
        }

        // ---- O += P * V (grouped B-fragment windows of 4) ----
        V16U pa[2];
        {
            const _Float16* pr = &pw[l16 * PSTR];
            pa[0].h[0] = *(const v8h*)(pr + 8 * g);        // k =  8g..8g+7
            pa[0].h[1] = *(const v8h*)(pr + 16 + 8 * g);   // k = 16+8g..
            pa[1].h[0] = *(const v8h*)(pr + 32 + 8 * g);   // k = 32+8g..
            pa[1].h[1] = *(const v8h*)(pr + 48 + 8 * g);   // k = 48+8g..
        }
#pragma unroll
        for (int dt2 = 0; dt2 < 4; ++dt2) {
            V16U bv[4];
#pragma unroll
            for (int i = 0; i < 4; ++i) {
                const int dt = dt2 * 2 + (i >> 1);
                const int kb = i & 1;
                const _Float16* vp = &sVt[(dt * 16 + l16) * VSTR + kb * 32 + g * 8];
                bv[i].h[0] = *(const v8h*)vp;
                bv[i].h[1] = *(const v8h*)(vp + 16);
            }
#pragma unroll
            for (int i = 0; i < 4; ++i) {
                const int dt = dt2 * 2 + (i >> 1);
                const int kb = i & 1;
                Oacc[dt] = __builtin_amdgcn_wmma_f32_16x16x32_f16(false, pa[kb].v, false, bv[i].v,
                                                                  (short)0, Oacc[dt], false, false);
            }
        }

        // ---- prefetch next K/V tile, then release LDS tiles ----
        if (kt + 1 < NKT) {
            const float* nk = K + base + (size_t)((kt + 1) * TK + srow) * DKK + scol;
            const float* nv = V + base + (size_t)((kt + 1) * TK + srow) * DKK + scol;
            __builtin_prefetch(nk, 0, 0);
            __builtin_prefetch(nv, 0, 0);
        }
        __syncthreads();
    }

    // ---- epilogue: O / l, write fp32 ----
    float* ob = Out + base;
#pragma unroll
    for (int vv = 0; vv < 8; ++vv) {
        const int r = vv + 8 * g;
        const float inv = 1.0f / lrow[vv];
        float* op = ob + (size_t)(q0 + r) * DKK + l16;
#pragma unroll
        for (int dt = 0; dt < 8; ++dt) op[dt * 16] = Oacc[dt][vv] * inv;
    }
}

extern "C" void kernel_launch(void* const* d_in, const int* in_sizes, int n_in,
                              void* d_out, int out_size, void* d_ws, size_t ws_size,
                              hipStream_t stream) {
    (void)in_sizes; (void)n_in; (void)out_size; (void)d_ws; (void)ws_size;
    const float* q = (const float*)d_in[0];
    const float* k = (const float*)d_in[1];
    const float* v = (const float*)d_in[2];
    float* out = (float*)d_out;
    // 16 batches x 16 query-blocks (128 rows each, 8 waves x 16 rows)
    dim3 grid(256), block(256);
    flash_attn_f16wmma<<<grid, block, 0, stream>>>(q, k, v, out);
}
